// GPT2Large_506806141199
// MI455X (gfx1250) — matrix operand order repaired and checked by reading, not compiled, not run on hardware
//
#include <hip/hip_runtime.h>
#include <cstdint>
#include <cstddef>

// ---------------- model dims ----------------
#define H      1280
#define NH     20
#define HD     64
#define LAYERS 36
#define SEQ    512
#define VOCAB  50257
#define FDIM   5120

typedef __attribute__((ext_vector_type(16))) __bf16 v16bf;
typedef __attribute__((ext_vector_type(8)))  float  v8f;

union Frag { v16bf v; unsigned int u[8]; };

__device__ __forceinline__ unsigned short f2bf(float f) {
  union { float f; unsigned int u; } c; c.f = f;
  unsigned int r = c.u + 0x7FFFu + ((c.u >> 16) & 1u);   // round-to-nearest-even
  return (unsigned short)(r >> 16);
}

// LDS-aperture flat pointers carry the allocation-relative byte offset in the
// low 32 bits (CDNA5 ISA 10.2) -> valid VDST payload for async-to-LDS ops.
__device__ __forceinline__ unsigned int lds_off(const void* p) {
  return (unsigned int)(unsigned long long)p;
}

__device__ __forceinline__ void async_copy16(unsigned int lds_addr,
                                             const void* gptr) {
  asm volatile("global_load_async_to_lds_b128 %0, %1, off"
               :: "v"(lds_addr), "v"((unsigned long long)gptr)
               : "memory");
}

__device__ __forceinline__ void wait_async() {
  asm volatile("s_wait_asynccnt 0" ::: "memory");
}

// =====================================================================
// Generic WMMA GEMM: C[M,N] = act(A_bf16[M,K] * B[K,N] (+bias) (+res))
//  - A: bf16 bits, row-major. Staged with global_load_async_to_lds_b128.
//  - B: TB = float  -> fp32 weights streamed once, cvt to bf16 in-register
//       TB = ushort -> bf16 activations; if BT, staged via async-to-LDS.
//    BT=true: B stored row-major [N,K] (compute A @ B^T).
//  - 256 threads = 8 wave32 waves; block tile 128x128, BK=32,
//    double-buffered LDS (global traffic for stage s+1 overlaps WMMA of s).
//  - blockIdx.z batches independent GEMMs (attention heads).
// =====================================================================
template<typename TB, bool BT, bool BIAS, bool GELU_ACT, bool RES, bool OUT_BF16>
__global__ __launch_bounds__(256)
void gemm_wmma(const unsigned short* __restrict__ A, int lda, long long bsA,
               const TB* __restrict__ B, int ldb, long long bsB,
               const float* __restrict__ bias,
               const float* __restrict__ resid, int ldr,
               void* __restrict__ Cout, int ldc, long long bsC,
               int M, int N, int K)
{
  constexpr bool B_F32    = (sizeof(TB) == 4);
  constexpr bool B_ASYNC  = (BT && !B_F32);   // bf16 [N,K]: layout-preserving
  __shared__ unsigned short As[2][128][34];   // [m][k] bf16, +2 pad
  __shared__ unsigned short Bs[2][128][34];   // [n][k] bf16 (frag-ready)

  const int tid   = threadIdx.x;
  const int lane  = tid & 31;
  const int wave  = tid >> 5;
  const int waveM = wave >> 1;              // 0..3
  const int waveN = wave & 1;               // 0..1
  const int tileM = blockIdx.y * 128;
  const int tileN = blockIdx.x * 128;
  const int batch = blockIdx.z;

  A += (long long)batch * bsA;
  B += (long long)batch * bsB;

  v8f acc[2][4] = {};

  float4 bregf[4];   // fp32 B prefetch registers
  uint4  bregu[2];   // bf16 [K,N] B prefetch registers

  // ---- stage A tile (always bf16, async direct-to-LDS) ----
  auto issueA = [&](int buf, int kk) {
#pragma unroll
    for (int p = 0; p < 2; ++p) {
      int m  = (tid >> 2) + p * 64;
      int k8 = (tid & 3) * 8;
      async_copy16(lds_off(&As[buf][m][k8]),
                   A + (long long)(tileM + m) * lda + kk + k8);
    }
  };

  // ---- B tile: issue global traffic (async or into regs) ----
  auto loadB = [&](int buf, int kk) {
    if constexpr (B_ASYNC) {                 // bf16 [N,K]: async, skip OOB rows
#pragma unroll
      for (int p = 0; p < 2; ++p) {
        int n  = (tid >> 2) + p * 64;
        int k8 = (tid & 3) * 8;
        int gn = tileN + n;
        if (gn < N)
          async_copy16(lds_off(&Bs[buf][n][k8]),
                       (const unsigned short*)B + (long long)gn * ldb + kk + k8);
      }
    } else if constexpr (BT && B_F32) {      // fp32 [N,K] (tied LM head)
#pragma unroll
      for (int p = 0; p < 4; ++p) {
        int n  = (tid >> 3) + p * 32;
        int kc = (tid & 7) * 4;
        int gn = tileN + n;
        bregf[p] = make_float4(0.f, 0.f, 0.f, 0.f);
        if (gn < N)
          bregf[p] = *(const float4*)((const float*)B + (long long)gn * ldb + kk + kc);
      }
    } else if constexpr (!BT && B_F32) {     // fp32 [K,N] weights
#pragma unroll
      for (int p = 0; p < 4; ++p) {
        int kr = (tid >> 5) + p * 8;
        int nc = (tid & 31) * 4;
        const float* src = (const float*)B + (long long)(kk + kr) * ldb + tileN + nc;
        if (tileN + nc + 3 < N) {
          bregf[p] = *(const float4*)src;
        } else {
          float v[4];
#pragma unroll
          for (int i = 0; i < 4; ++i) v[i] = (tileN + nc + i < N) ? src[i] : 0.f;
          bregf[p] = make_float4(v[0], v[1], v[2], v[3]);
        }
      }
    } else {                                 // bf16 [K,N] (attention V)
#pragma unroll
      for (int p = 0; p < 2; ++p) {
        int kr = (tid >> 4) + p * 16;
        int nc = (tid & 15) * 8;
        const unsigned short* src =
            (const unsigned short*)B + (long long)(kk + kr) * ldb + tileN + nc;
        if (tileN + nc + 7 < N) {
          bregu[p] = *(const uint4*)src;
        } else {
          unsigned short v[8];
#pragma unroll
          for (int i = 0; i < 8; ++i)
            v[i] = (tileN + nc + i < N) ? src[i] : (unsigned short)0;
          bregu[p] = *(const uint4*)v;
        }
      }
    }
  };

  // ---- B tile: commit register-staged data to LDS (post-compute) ----
  auto storeB = [&](int buf) {
    if constexpr (B_ASYNC) {
      // nothing: async engine wrote LDS directly
    } else if constexpr (BT && B_F32) {
#pragma unroll
      for (int p = 0; p < 4; ++p) {
        int n  = (tid >> 3) + p * 32;
        int kc = (tid & 7) * 4;
        *(unsigned int*)&Bs[buf][n][kc + 0] =
            ((unsigned int)f2bf(bregf[p].y) << 16) | f2bf(bregf[p].x);
        *(unsigned int*)&Bs[buf][n][kc + 2] =
            ((unsigned int)f2bf(bregf[p].w) << 16) | f2bf(bregf[p].z);
      }
    } else if constexpr (!BT && B_F32) {
#pragma unroll
      for (int p = 0; p < 4; ++p) {
        int kr = (tid >> 5) + p * 8;
        int nc = (tid & 31) * 4;
        Bs[buf][nc + 0][kr] = f2bf(bregf[p].x);
        Bs[buf][nc + 1][kr] = f2bf(bregf[p].y);
        Bs[buf][nc + 2][kr] = f2bf(bregf[p].z);
        Bs[buf][nc + 3][kr] = f2bf(bregf[p].w);
      }
    } else {
#pragma unroll
      for (int p = 0; p < 2; ++p) {
        int kr = (tid >> 4) + p * 16;
        int nc = (tid & 15) * 8;
        unsigned short v[8];
        *(uint4*)v = bregu[p];
#pragma unroll
        for (int i = 0; i < 8; ++i) Bs[buf][nc + i][kr] = v[i];
      }
    }
  };

  // ---- fragments (documented 16-bit layouts) + 8 WMMAs ----
  auto compute = [&](int buf) {
    Frag af[2];
#pragma unroll
    for (int mi = 0; mi < 2; ++mi) {
      int mrow = waveM * 32 + mi * 16 + (lane & 15);
      int kb   = (lane >> 4) * 8;
#pragma unroll
      for (int j = 0; j < 8; ++j) {
        int k = ((j >> 2) * 16) + kb + ((j & 3) * 2);
        af[mi].u[j] = *(const unsigned int*)&As[buf][mrow][k];
      }
    }
#pragma unroll
    for (int ni = 0; ni < 4; ++ni) {
      Frag bfr;
      int nrow = waveN * 64 + ni * 16 + (lane & 15);
      int kb   = (lane >> 4) * 8;
#pragma unroll
      for (int j = 0; j < 8; ++j)
        bfr.u[j] = *(const unsigned int*)&Bs[buf][nrow][((j >> 2) * 16) + kb + ((j & 3) * 2)];
#pragma unroll
      for (int mi = 0; mi < 2; ++mi)
        acc[mi][ni] = __builtin_amdgcn_wmma_f32_16x16x32_bf16(
            false, af[mi].v, false, bfr.v, (short)0, acc[mi][ni], false, false);
    }
  };

  // ---- double-buffered main loop ----
  const int nsteps = K >> 5;   // K is a multiple of 32 for every GEMM here
  issueA(0, 0);
  loadB(0, 0);
  storeB(0);
  wait_async();
  __syncthreads();

  for (int s = 0; s < nsteps; ++s) {
    const int buf = s & 1;
    const int nxt = buf ^ 1;
    if (s + 1 < nsteps) {
      issueA(nxt, (s + 1) << 5);     // async: lands in LDS while we compute
      loadB(nxt, (s + 1) << 5);      // reg-staged: loads in flight over WMMA
    }
    compute(buf);
    if (s + 1 < nsteps) storeB(nxt);
    wait_async();                     // ASYNCcnt not covered by __syncthreads
    __syncthreads();
  }

  // ---- epilogue: C/D f32 layout: VGPR i -> M = i + (L>=16 ? 8 : 0) ----
  const int rbase = tileM + waveM * 32 + ((lane >> 4) << 3);
  const int col0  = tileN + waveN * 64 + (lane & 15);
#pragma unroll
  for (int mi = 0; mi < 2; ++mi) {
#pragma unroll
    for (int ni = 0; ni < 4; ++ni) {
      int col = col0 + ni * 16;
      if (col >= N) continue;
      float bcol = 0.0f;
      if constexpr (BIAS) bcol = bias[col];
#pragma unroll
      for (int i = 0; i < 8; ++i) {
        int row = rbase + mi * 16 + i;
        if (row >= M) continue;
        float v = acc[mi][ni][i];
        if constexpr (BIAS)     v += bcol;
        if constexpr (GELU_ACT) v = 0.5f * v * (1.0f + erff(v * 0.70710678118654752f));
        if constexpr (RES)      v += resid[(long long)row * ldr + col];
        long long off = (long long)batch * bsC + (long long)row * ldc + col;
        if constexpr (OUT_BF16) ((unsigned short*)Cout)[off] = f2bf(v);
        else                    ((float*)Cout)[off] = v;
      }
    }
  }
}

// =====================================================================
// Embedding gather: x[s,:] = wte[ids[s],:] + wpe[s,:]
// =====================================================================
__global__ __launch_bounds__(256)
void embed_kernel(const int* __restrict__ ids, const float* __restrict__ wte,
                  const float* __restrict__ wpe, float* __restrict__ x)
{
  int s = blockIdx.x;
  long long id = ids[s];
  for (int c = threadIdx.x; c < H; c += 256)
    x[(long long)s * H + c] = wte[id * H + c] + wpe[(long long)s * H + c];
}

// =====================================================================
// LayerNorm fp32 -> bf16 (H = 1280 = 256 * 5)
// =====================================================================
__global__ __launch_bounds__(256)
void ln_bf16_kernel(const float* __restrict__ x, const float* __restrict__ w,
                    const float* __restrict__ b, unsigned short* __restrict__ out)
{
  int row = blockIdx.x, t = threadIdx.x;
  const float* xr = x + (long long)row * H;
  float vals[5], s = 0.f, s2 = 0.f;
#pragma unroll
  for (int i = 0; i < 5; ++i) {
    float v = xr[t + i * 256];
    vals[i] = v; s += v; s2 += v * v;
  }
  __shared__ float rs[256], rq[256];
  rs[t] = s; rq[t] = s2; __syncthreads();
  for (int off = 128; off > 0; off >>= 1) {
    if (t < off) { rs[t] += rs[t + off]; rq[t] += rq[t + off]; }
    __syncthreads();
  }
  float mean = rs[0] * (1.0f / H);
  float var  = rq[0] * (1.0f / H) - mean * mean;
  float inv  = rsqrtf(var + 1e-5f);
#pragma unroll
  for (int i = 0; i < 5; ++i) {
    int c = t + i * 256;
    out[(long long)row * H + c] = f2bf((vals[i] - mean) * inv * w[c] + b[c]);
  }
}

// =====================================================================
// Causal scaled softmax over one row of scores -> bf16 probs
// blockIdx.x = h*SEQ + q ;  512 cols, 2 per thread
// =====================================================================
__global__ __launch_bounds__(256)
void softmax_causal_kernel(const float* __restrict__ scores,
                           unsigned short* __restrict__ probs)
{
  const int row = blockIdx.x;
  const int q   = row & (SEQ - 1);
  const float scale = 0.125f;                 // 1/sqrt(HD)
  const float* sr = scores + (long long)row * SEQ;
  unsigned short* pr = probs + (long long)row * SEQ;
  int t = threadIdx.x;
  float v[2];
#pragma unroll
  for (int i = 0; i < 2; ++i) {
    int c = t + i * 256;
    v[i] = (c <= q) ? sr[c] * scale : -1e30f;
  }
  __shared__ float red[256];
  red[t] = fmaxf(v[0], v[1]); __syncthreads();
  for (int off = 128; off > 0; off >>= 1) {
    if (t < off) red[t] = fmaxf(red[t], red[t + off]);
    __syncthreads();
  }
  float m = red[0]; __syncthreads();
  float e[2]; float s = 0.f;
#pragma unroll
  for (int i = 0; i < 2; ++i) { e[i] = __expf(v[i] - m); s += e[i]; }
  red[t] = s; __syncthreads();
  for (int off = 128; off > 0; off >>= 1) {
    if (t < off) red[t] += red[t + off];
    __syncthreads();
  }
  float inv = 1.0f / red[0];
#pragma unroll
  for (int i = 0; i < 2; ++i) pr[t + i * 256] = f2bf(e[i] * inv);
}

// =====================================================================
// Host orchestration
// =====================================================================
static inline size_t alignup(size_t v) { return (v + 255) & ~(size_t)255; }

extern "C" void kernel_launch(void* const* d_in, const int* in_sizes, int n_in,
                              void* d_out, int out_size, void* d_ws, size_t ws_size,
                              hipStream_t stream)
{
  (void)in_sizes; (void)n_in; (void)out_size; (void)ws_size;

  const int*   ids     = (const int*)  d_in[0];
  const float* wte     = (const float*)d_in[1];
  const float* wpe     = (const float*)d_in[2];
  const float* ln1_w   = (const float*)d_in[3];
  const float* ln1_b   = (const float*)d_in[4];
  const float* attn_w  = (const float*)d_in[5];
  const float* attn_b  = (const float*)d_in[6];
  const float* projA_w = (const float*)d_in[7];
  const float* projA_b = (const float*)d_in[8];
  const float* ln2_w   = (const float*)d_in[9];
  const float* ln2_b   = (const float*)d_in[10];
  const float* fc_w    = (const float*)d_in[11];
  const float* fc_b    = (const float*)d_in[12];
  const float* projM_w = (const float*)d_in[13];
  const float* projM_b = (const float*)d_in[14];
  const float* lnf_w   = (const float*)d_in[15];
  const float* lnf_b   = (const float*)d_in[16];

  // ---- workspace carve-up (~46 MB) ----
  char* p = (char*)d_ws;
  float*          x      = (float*)p;          p += alignup((size_t)SEQ * H * 4);
  unsigned short* lnout  = (unsigned short*)p; p += alignup((size_t)SEQ * H * 2);
  unsigned short* qkv    = (unsigned short*)p; p += alignup((size_t)SEQ * 3 * H * 2);
  float*          scores = (float*)p;          p += alignup((size_t)NH * SEQ * SEQ * 4);
  unsigned short* probs  = (unsigned short*)p; p += alignup((size_t)NH * SEQ * SEQ * 2);
  unsigned short* ctx    = (unsigned short*)p; p += alignup((size_t)SEQ * H * 2);
  unsigned short* fcbuf  = (unsigned short*)p; p += alignup((size_t)SEQ * FDIM * 2);

  const dim3 blk(256);

  // embeddings
  embed_kernel<<<SEQ, blk, 0, stream>>>(ids, wte, wpe, x);

  for (int l = 0; l < LAYERS; ++l) {
    const float* aw = attn_w  + (size_t)l * H * 3 * H;
    const float* ab = attn_b  + (size_t)l * 3 * H;
    const float* pw = projA_w + (size_t)l * H * H;
    const float* pb = projA_b + (size_t)l * H;
    const float* fw = fc_w    + (size_t)l * H * FDIM;
    const float* fb = fc_b    + (size_t)l * FDIM;
    const float* mw = projM_w + (size_t)l * FDIM * H;
    const float* mb = projM_b + (size_t)l * H;

    // LN1 -> bf16
    ln_bf16_kernel<<<SEQ, blk, 0, stream>>>(x, ln1_w + (size_t)l * H,
                                            ln1_b + (size_t)l * H, lnout);

    // QKV = lnout @ attn_w + attn_b  -> bf16 [S, 3H]
    gemm_wmma<float, false, true, false, false, true>
        <<<dim3(3 * H / 128, SEQ / 128, 1), blk, 0, stream>>>(
        lnout, H, 0, aw, 3 * H, 0, ab, nullptr, 0,
        qkv, 3 * H, 0, SEQ, 3 * H, H);

    // scores[h] = q_h @ k_h^T  -> f32 [NH, S, S]  (batched over heads)
    gemm_wmma<unsigned short, true, false, false, false, false>
        <<<dim3(SEQ / 128, SEQ / 128, NH), blk, 0, stream>>>(
        qkv, 3 * H, HD, qkv + H, 3 * H, HD, nullptr, nullptr, 0,
        scores, SEQ, (long long)SEQ * SEQ, SEQ, SEQ, HD);

    // causal softmax -> bf16 probs
    softmax_causal_kernel<<<NH * SEQ, blk, 0, stream>>>(scores, probs);

    // ctx[h] = probs_h @ v_h  -> bf16 [S, H] (head-sliced, batched)
    gemm_wmma<unsigned short, false, false, false, false, true>
        <<<dim3(1, SEQ / 128, NH), blk, 0, stream>>>(
        probs, SEQ, (long long)SEQ * SEQ, qkv + 2 * H, 3 * H, HD,
        nullptr, nullptr, 0, ctx, H, HD, SEQ, HD, SEQ);

    // x = x + ctx @ projA_w + projA_b   (f32, in-place residual)
    gemm_wmma<float, false, true, false, true, false>
        <<<dim3(H / 128, SEQ / 128, 1), blk, 0, stream>>>(
        ctx, H, 0, pw, H, 0, pb, x, H, x, H, 0, SEQ, H, H);

    // LN2 -> bf16
    ln_bf16_kernel<<<SEQ, blk, 0, stream>>>(x, ln2_w + (size_t)l * H,
                                            ln2_b + (size_t)l * H, lnout);

    // fc = gelu(lnout @ fc_w + fc_b) -> bf16 [S, F]
    gemm_wmma<float, false, true, true, false, true>
        <<<dim3(FDIM / 128, SEQ / 128, 1), blk, 0, stream>>>(
        lnout, H, 0, fw, FDIM, 0, fb, nullptr, 0,
        fcbuf, FDIM, 0, SEQ, FDIM, H);

    // x = x + fc @ projM_w + projM_b
    gemm_wmma<float, false, true, false, true, false>
        <<<dim3(H / 128, SEQ / 128, 1), blk, 0, stream>>>(
        fcbuf, FDIM, 0, mw, H, 0, mb, x, H, x, H, 0, SEQ, H, FDIM);
  }

  // final LN -> bf16
  ln_bf16_kernel<<<SEQ, blk, 0, stream>>>(x, lnf_w, lnf_b, lnout);

  // logits = lnout @ wte^T  -> f32 [S, V]
  gemm_wmma<float, true, false, false, false, false>
      <<<dim3((VOCAB + 127) / 128, SEQ / 128, 1), blk, 0, stream>>>(
      lnout, H, 0, wte, H, 0, nullptr, nullptr, 0,
      (float*)d_out, VOCAB, 0, SEQ, VOCAB, H);
}
